// NonLocalBlock_53051436040480
// MI455X (gfx1250) — matrix operand order repaired and checked by reading, not compiled, not run on hardware
//
#include <hip/hip_runtime.h>
#include <hip/hip_bf16.h>

// ---------------------------------------------------------------------------
// NonLocal block (self-attention) for MI455X / gfx1250, wave32 + WMMA.
//   B=8, C=128, T=4096, H=64
// Pipeline:
//   1) proj:    theta/phi  -> f16 [B][T][H]   (h contiguous)
//               g          -> f16 [B][H][T]   (t contiguous)
//   2) attn:    flash-style fused scores+softmax+PV, 64-wide s chunks,
//               O -> f16 [B][T][H]
//   3) outproj: y = x + w_out @ O + b_out     (f32 output)
// All heavy math uses v_wmma_f32_16x16x32_f16 (f32 accumulate).
// ---------------------------------------------------------------------------

typedef __attribute__((ext_vector_type(16))) _Float16 v16h;
typedef __attribute__((ext_vector_type(8)))  _Float16 v8h;
typedef __attribute__((ext_vector_type(8)))  float    v8f;
typedef __attribute__((ext_vector_type(4)))  float    v4f;

namespace {
constexpr int BB = 8;     // batch
constexpr int CC = 128;   // channels
constexpr int TT = 4096;  // sequence
constexpr int HH = 64;    // hidden
}

#define WMMA32(A, Bm, Cacc) \
  __builtin_amdgcn_wmma_f32_16x16x32_f16(false, (A), false, (Bm), (short)0, (Cacc), false, false)

__device__ __forceinline__ v8f zero8() {
  v8f z;
#pragma unroll
  for (int i = 0; i < 8; ++i) z[i] = 0.0f;
  return z;
}

// A-fragment (16x32 f16, M x K): lane = row (lane&15); elements 0..15 hold
// K = base+khiA+0..15, with
// khiA = (lane>=16 ? 16 : 0) already folded into the pointer by the caller.
__device__ __forceinline__ v16h ld_a_h(const _Float16* __restrict__ p) {
  v8h lo = *(const v8h*)(p);
  v8h hi = *(const v8h*)(p + 8);
  v16h r;
#pragma unroll
  for (int i = 0; i < 8; ++i) { r[i] = lo[i]; r[i + 8] = hi[i]; }
  return r;
}

__device__ __forceinline__ v16h ld_a_f32(const float* __restrict__ p) {
  v16h r;
#pragma unroll
  for (int i = 0; i < 8; ++i) {
    r[i]     = (_Float16)p[i];
    r[i + 8] = (_Float16)p[8 + i];
  }
  return r;
}

// B-fragment (32x16 f16, K x N): lane = col (lane&15); elements 0..15 hold
// K = base+khiB+0..15 with khiB = (lane>=16 ? 16 : 0) folded in by caller.
__device__ __forceinline__ v16h ld_b_f32(const float* __restrict__ p) {
  v16h r;
#pragma unroll
  for (int i = 0; i < 16; ++i) r[i] = (_Float16)p[i];
  return r;
}

__device__ __forceinline__ float redmax16(float v) {
  v = fmaxf(v, __shfl_xor(v, 1, 32));
  v = fmaxf(v, __shfl_xor(v, 2, 32));
  v = fmaxf(v, __shfl_xor(v, 4, 32));
  v = fmaxf(v, __shfl_xor(v, 8, 32));
  return v;
}
__device__ __forceinline__ float redsum16(float v) {
  v += __shfl_xor(v, 1, 32);
  v += __shfl_xor(v, 2, 32);
  v += __shfl_xor(v, 4, 32);
  v += __shfl_xor(v, 8, 32);
  return v;
}

// ---------------------------------------------------------------------------
// Kernel 1: projections.  grid = (TT/128, BB), block = 256 (8 waves).
// theta[b,h,t] = sum_c w_theta[h,c] x[b,c,t] + b_theta[h]   (same for phi)
// g stored transposed as [B][H][T].
// Tile work: theta: 4(h) x 8(t) tiles; phi: same; g: 8(t) x 4(h) tiles.
// ---------------------------------------------------------------------------
__global__ void __launch_bounds__(256)
nl_proj_kernel(const float* __restrict__ x,
               const float* __restrict__ w_theta, const float* __restrict__ b_theta,
               const float* __restrict__ w_phi,   const float* __restrict__ b_phi,
               const float* __restrict__ w_g,     const float* __restrict__ b_g,
               _Float16* __restrict__ theta16,
               _Float16* __restrict__ phi16,
               _Float16* __restrict__ g16) {
  const int b    = blockIdx.y;
  const int t0   = blockIdx.x * 128;
  const int tid  = threadIdx.x;
  const int lane = tid & 31;
  const int wave = tid >> 5;

  // x tile transposed to f16: xT[t][c], padded row (+8) for bank rotation.
  __shared__ _Float16 xT[128][136];

  {
    const float* xb = x + (size_t)b * CC * TT;
#pragma unroll
    for (int i = tid; i < 128 * 128 / 4; i += 256) {
      const int c  = i >> 5;          // 0..127
      const int tq = (i & 31) * 4;    // 0..124
      v4f v = *(const v4f*)(xb + (size_t)c * TT + t0 + tq);
#pragma unroll
      for (int j = 0; j < 4; ++j) xT[tq + j][c] = (_Float16)v[j];
    }
  }
  __syncthreads();

  const int col  = lane & 15;
  const int khiA = (lane & 16) ? 16 : 0;
  const int khiB = (lane & 16) ? 16 : 0;
  const int roff = (lane & 16) ? 8 : 0;

  for (int tile = wave; tile < 96; tile += 8) {
    const int mat = tile >> 5;   // 0: theta, 1: phi, 2: g
    const int idx = tile & 31;
    v8f acc = zero8();

    if (mat < 2) {
      const float*    W    = (mat == 0) ? w_theta : w_phi;
      const float*    bias = (mat == 0) ? b_theta : b_phi;
      _Float16*       out  = (mat == 0) ? theta16 : phi16;
      const int mt = idx >> 3;       // h tile
      const int nt = idx & 7;        // t tile
      const int arow = mt * 16 + col;
      const int bcol = nt * 16 + col;
#pragma unroll
      for (int ks = 0; ks < 4; ++ks) {
        v16h Af = ld_a_f32(W + (size_t)arow * CC + ks * 32 + khiA);
        v16h Bf = *(const v16h*)&xT[bcol][ks * 32 + khiB];
        acc = WMMA32(Af, Bf, acc);
      }
      // D: col = t (lane), rows = h = mt*16 + roff + r  (8 contiguous h)
      const int t     = t0 + bcol;
      const int hbase = mt * 16 + roff;
      v8h o;
#pragma unroll
      for (int r = 0; r < 8; ++r) o[r] = (_Float16)(acc[r] + bias[hbase + r]);
      *(v8h*)(out + ((size_t)b * TT + t) * HH + hbase) = o;
    } else {
      // g^T: out[t,h] = sum_c xT[t,c] * w_g[h,c]
      const int mt = idx >> 2;       // t tile
      const int nt = idx & 3;        // h tile
      const int arow = mt * 16 + col;   // t row in xT
      const int bcol = nt * 16 + col;   // h col
#pragma unroll
      for (int ks = 0; ks < 4; ++ks) {
        v16h Af = ld_a_h(&xT[arow][0] + ks * 32 + khiA);
        v16h Bf = ld_b_f32(w_g + (size_t)bcol * CC + ks * 32 + khiB);
        acc = WMMA32(Af, Bf, acc);
      }
      // D: col = h (lane), rows = t (8 contiguous t) -> g16[b][h][t]
      const int h    = bcol;
      const int trow = t0 + mt * 16 + roff;
      const float bg = b_g[h];
      v8h o;
#pragma unroll
      for (int r = 0; r < 8; ++r) o[r] = (_Float16)(acc[r] + bg);
      *(v8h*)(g16 + ((size_t)b * HH + h) * TT + trow) = o;
    }
  }
}

// ---------------------------------------------------------------------------
// Kernel 2: fused attention (flash style).  grid = (TT/128, BB), block = 256.
// Each wave owns a 16-row t tile; block streams 64-wide s chunks (amortizes
// the online-softmax shuffles/rescales over twice the WMMA work).
//   S = theta_tile^T @ phi_chunk   (K = H = 64 -> 2 WMMA per 16x16 tile, x4)
//   online softmax over s, P -> per-wave LDS tile -> two A-fragments (K=64)
//   O += P @ g_chunk^T             (4 h-tiles x 2 K-steps = 8 WMMA)
// ---------------------------------------------------------------------------
__global__ void __launch_bounds__(256)
nl_attn_kernel(const _Float16* __restrict__ theta16,
               const _Float16* __restrict__ phi16,
               const _Float16* __restrict__ g16,
               _Float16* __restrict__ O16) {
  const int b    = blockIdx.y;
  const int t0   = blockIdx.x * 128;
  const int tid  = threadIdx.x;
  const int lane = tid & 31;
  const int wave = tid >> 5;

  __shared__ _Float16 sPhi[2][64][72];  // [s][h], padded
  __shared__ _Float16 sG[2][64][72];    // [h][s], padded
  __shared__ _Float16 sP[8][16][72];    // per-wave P tile [t][s], padded

  const int col  = lane & 15;
  const int khiA = (lane & 16) ? 16 : 0;
  const int khiB = (lane & 16) ? 16 : 0;
  const int roff = (lane & 16) ? 8 : 0;

  // Per-wave query fragments (K = 64 -> two A fragments), kept in VGPRs.
  const int trow = t0 + wave * 16 + col;
  const _Float16* throw_p = theta16 + ((size_t)b * TT + trow) * HH;
  const v16h A0 = ld_a_h(throw_p + 0 + khiA);    // h 0..31
  const v16h A1 = ld_a_h(throw_p + 32 + khiA);   // h 32..63

  v8f o[4];
#pragma unroll
  for (int n = 0; n < 4; ++n) o[n] = zero8();
  float mrow[8], lrow[8];
#pragma unroll
  for (int r = 0; r < 8; ++r) { mrow[r] = -3.0e38f; lrow[r] = 0.0f; }

  const _Float16* phiB = phi16 + (size_t)b * TT * HH;
  const _Float16* gB   = g16 + (size_t)b * HH * TT;

  // phi chunk: 64 rows x 64 halves; g chunk: 64 rows x 64 halves.
  // 256 threads, 16 halves (32B) per thread per matrix.
  const int pr = tid >> 2;          // 0..63
  const int pc = (tid & 3) * 16;    // 0, 16, 32, 48

  auto load_chunk = [&](int s0, int buf) {
    *(v16h*)&sPhi[buf][pr][pc] = *(const v16h*)(phiB + (size_t)(s0 + pr) * HH + pc);
    *(v16h*)&sG[buf][pr][pc]   = *(const v16h*)(gB + (size_t)pr * TT + s0 + pc);
  };

  load_chunk(0, 0);
  __syncthreads();

  for (int it = 0; it < TT / 64; ++it) {
    const int buf = it & 1;
    const int s0n = (it + 1) * 64;
    if (s0n < TT) {
      load_chunk(s0n, buf ^ 1);
      // prefetch two chunks ahead into GL2 (global_prefetch_b8)
      if (s0n + 64 < TT) {
        __builtin_prefetch(phiB + (size_t)(s0n + 64 + pr) * HH + pc, 0, 1);
        __builtin_prefetch(gB + (size_t)pr * TT + s0n + 64 + pc, 0, 1);
      }
    }

    // ---- scores for this 64-wide s chunk: four 16x16 tiles ----------------
    v8f sa[4];
#pragma unroll
    for (int nt = 0; nt < 4; ++nt) {
      v16h B0 = *(const v16h*)&sPhi[buf][nt * 16 + col][0 + khiB];
      v16h B1 = *(const v16h*)&sPhi[buf][nt * 16 + col][32 + khiB];
      v8f acc = zero8();
      acc = WMMA32(A0, B0, acc);
      acc = WMMA32(A1, B1, acc);
      sa[nt] = acc;
    }

    // ---- online softmax (row-wise; row r lives on one 16-lane half) -------
#pragma unroll
    for (int r = 0; r < 8; ++r) {
      const float v0 = sa[0][r], v1 = sa[1][r], v2 = sa[2][r], v3 = sa[3][r];
      const float cmax = redmax16(fmaxf(fmaxf(v0, v1), fmaxf(v2, v3)));
      const float mnew = fmaxf(mrow[r], cmax);
      const float alpha = __expf(mrow[r] - mnew);
      mrow[r] = mnew;
      const float p0 = __expf(v0 - mnew);
      const float p1 = __expf(v1 - mnew);
      const float p2 = __expf(v2 - mnew);
      const float p3 = __expf(v3 - mnew);
      lrow[r] = lrow[r] * alpha + redsum16((p0 + p1) + (p2 + p3));
#pragma unroll
      for (int n = 0; n < 4; ++n) o[n][r] *= alpha;
      const int prow = r + roff;
      sP[wave][prow][col]      = (_Float16)p0;
      sP[wave][prow][16 + col] = (_Float16)p1;
      sP[wave][prow][32 + col] = (_Float16)p2;
      sP[wave][prow][48 + col] = (_Float16)p3;
    }

    // ---- O += P @ g^T (wave-private P tile; in-wave DS ordering) ----------
    const v16h Ap0 = ld_a_h(&sP[wave][col][0] + khiA);        // s 0..31
    const v16h Ap1 = ld_a_h(&sP[wave][col][0] + 32 + khiA);   // s 32..63
#pragma unroll
    for (int n = 0; n < 4; ++n) {
      v16h Bg0 = *(const v16h*)&sG[buf][n * 16 + col][0 + khiB];
      v16h Bg1 = *(const v16h*)&sG[buf][n * 16 + col][32 + khiB];
      o[n] = WMMA32(Ap0, Bg0, o[n]);
      o[n] = WMMA32(Ap1, Bg1, o[n]);
    }

    __syncthreads();   // next chunk ready; safe to overwrite buf next iter
  }

  // ---- finalize: O /= l, store as f16 [B][T][H] ---------------------------
  float inv[8];
#pragma unroll
  for (int r = 0; r < 8; ++r) inv[r] = 1.0f / lrow[r];

  const int tbase = t0 + wave * 16 + roff;
#pragma unroll
  for (int n = 0; n < 4; ++n) {
    const int h = n * 16 + col;
#pragma unroll
    for (int r = 0; r < 8; ++r) {
      O16[((size_t)b * TT + (tbase + r)) * HH + h] = (_Float16)(o[n][r] * inv[r]);
    }
  }
}

// ---------------------------------------------------------------------------
// Kernel 3: y = x + w_out @ O + b_out.  grid = (TT/128, BB), block = 256.
// Each wave owns a 16-wide t tile and walks all 8 c tiles (K = H = 64).
// ---------------------------------------------------------------------------
__global__ void __launch_bounds__(256)
nl_outproj_kernel(const float* __restrict__ x,
                  const float* __restrict__ w_out,
                  const float* __restrict__ b_out,
                  const _Float16* __restrict__ O16,
                  float* __restrict__ y) {
  const int b    = blockIdx.y;
  const int t0   = blockIdx.x * 128;
  const int tid  = threadIdx.x;
  const int lane = tid & 31;
  const int wave = tid >> 5;

  const int col  = lane & 15;
  const int khiA = (lane & 16) ? 16 : 0;
  const int khiB = (lane & 16) ? 16 : 0;
  const int roff = (lane & 16) ? 8 : 0;

  const int tcol = t0 + wave * 16 + col;
  const _Float16* orow = O16 + ((size_t)b * TT + tcol) * HH;
  const v16h B0 = *(const v16h*)(orow + 0 + khiB);    // h 0..31
  const v16h B1 = *(const v16h*)(orow + 32 + khiB);   // h 32..63

#pragma unroll
  for (int mt = 0; mt < 8; ++mt) {
    const int crow = mt * 16 + col;
    v16h A0 = ld_a_f32(w_out + (size_t)crow * HH + 0 + khiA);
    v16h A1 = ld_a_f32(w_out + (size_t)crow * HH + 32 + khiA);
    v8f acc = zero8();
    acc = WMMA32(A0, B0, acc);
    acc = WMMA32(A1, B1, acc);
    // D: col = t (lane), rows = c = mt*16 + roff + r
#pragma unroll
    for (int r = 0; r < 8; ++r) {
      const int c = mt * 16 + roff + r;
      const size_t off = ((size_t)b * CC + c) * TT + tcol;
      y[off] = x[off] + acc[r] + b_out[c];
    }
  }
}

// ---------------------------------------------------------------------------
extern "C" void kernel_launch(void* const* d_in, const int* in_sizes, int n_in,
                              void* d_out, int out_size, void* d_ws, size_t ws_size,
                              hipStream_t stream) {
  (void)in_sizes; (void)n_in; (void)out_size; (void)ws_size;

  const float* x       = (const float*)d_in[0];
  const float* w_theta = (const float*)d_in[1];
  const float* b_theta = (const float*)d_in[2];
  const float* w_phi   = (const float*)d_in[3];
  const float* b_phi   = (const float*)d_in[4];
  const float* w_g     = (const float*)d_in[5];
  const float* b_g     = (const float*)d_in[6];
  const float* w_out   = (const float*)d_in[7];
  const float* b_out   = (const float*)d_in[8];
  float* y = (float*)d_out;

  // Workspace layout (f16): theta, phi [B][T][H]; g [B][H][T]; O [B][T][H].
  const size_t nBTH = (size_t)BB * TT * HH;
  _Float16* theta16 = (_Float16*)d_ws;
  _Float16* phi16   = theta16 + nBTH;
  _Float16* g16     = phi16 + nBTH;
  _Float16* O16     = g16 + nBTH;

  const dim3 grid(TT / 128, BB);
  const dim3 block(256);

  nl_proj_kernel<<<grid, block, 0, stream>>>(x, w_theta, b_theta, w_phi, b_phi,
                                             w_g, b_g, theta16, phi16, g16);
  nl_attn_kernel<<<grid, block, 0, stream>>>(theta16, phi16, g16, O16);
  nl_outproj_kernel<<<grid, block, 0, stream>>>(x, w_out, b_out, O16, y);
}
